// SeqCausalNet_86234353369167
// MI455X (gfx1250) — matrix operand order repaired and checked by reading, not compiled, not run on hardware
//
#include <hip/hip_runtime.h>
#include <hip/hip_bf16.h>

// ---------------------------------------------------------------------------
// MI455X (gfx1250) fused implementation of SeqCausalNet forward.
//
// Key simplifications proven from the reference math:
//  * inputs @ W_e1 is a gather: hid = relu(val*W_e1[0] + W_e1[f+1]*valid + b_e1)
//  * attention is permutation invariant -> the argsort is unnecessary;
//    token t == feature t, valid iff acquired[b,t] != 0.
//  * invalid tokens get softmax weight exp(-1e9 - max) == 0.0f exactly,
//    lengths==0 rows are force-zeroed, matching the reference.
//
// All GEMM-shaped work runs on v_wmma_f32_16x16x32_bf16 (bf16 in, f32 acc).
// ---------------------------------------------------------------------------

#define BATCH 2048
#define NFEAT 256
#define EH    128
#define EDIM  64
#define HDIM  64
#define SHID  256
#define SDIM  128
#define NA    257
#define NAPAD 272   // 17 * 16

typedef __attribute__((ext_vector_type(16))) __bf16 v16bf;
typedef __attribute__((ext_vector_type(8)))  float  v8f;

// ---- bf16 <-> f32 via raw bits (storage type: unsigned short) -------------
__device__ __forceinline__ unsigned short f2bf(float f) {
    unsigned int u = __float_as_uint(f);
    unsigned int r = u + 0x7FFFu + ((u >> 16) & 1u);   // round-to-nearest-even
    return (unsigned short)(r >> 16);
}
__device__ __forceinline__ float bf2f(unsigned short h) {
    return __uint_as_float(((unsigned int)h) << 16);
}

// ---- WMMA fragment loaders -------------------------------------------------
// A operand (16x32 bf16, M x K): lane m (0-15) holds row m, K in {kb+0..7, kb+16..23};
// lanes 16-31 hold the +8 shifted halves.  Two 16-byte chunks per lane.
__device__ __forceinline__ v16bf load_frag_a(const unsigned short* base, int ld,
                                             int row0, int kb, int lane) {
    const unsigned short* p = base + (row0 + (lane & 15)) * ld + kb + ((lane & 16) ? 8 : 0);
    union { uint4 q[2]; v16bf v; } u;
    u.q[0] = *(const uint4*)(p);
    u.q[1] = *(const uint4*)(p + 16);
    return u.v;
}
// B operand (32x16 bf16, K x N) sourced from W^T stored [N,K] row-major:
// lane n (0-15) holds column n, K in {kb+0..7, kb+16..23}; lanes 16-31 hold the +8 shifted halves.
__device__ __forceinline__ v16bf load_frag_b(const unsigned short* base, int ld,
                                             int n0, int kb, int lane) {
    const unsigned short* p = base + (n0 + (lane & 15)) * ld + kb + ((lane & 16) ? 8 : 0);
    union { uint4 q[2]; v16bf v; } u;
    u.q[0] = *(const uint4*)(p);
    u.q[1] = *(const uint4*)(p + 16);
    return u.v;
}

__device__ __forceinline__ float sigmoidf(float x) { return 1.0f / (1.0f + __expf(-x)); }

// ---- block reductions ------------------------------------------------------
__device__ __forceinline__ float block_sum256(float* red, int tid, float v) {
    red[tid] = v; __syncthreads();
    for (int s = 128; s >= 1; s >>= 1) {
        if (tid < s) red[tid] += red[tid + s];
        __syncthreads();
    }
    float r = red[0]; __syncthreads();
    return r;
}
__device__ __forceinline__ float block_max256(float* red, int tid, float v) {
    red[tid] = v; __syncthreads();
    for (int s = 128; s >= 1; s >>= 1) {
        if (tid < s) red[tid] = fmaxf(red[tid], red[tid + s]);
        __syncthreads();
    }
    float r = red[0]; __syncthreads();
    return r;
}

// ---------------------------------------------------------------------------
// Weight prep: dst_bf16[Npad][K] = transpose(src_f32[K][N]), zero-padded cols.
// ---------------------------------------------------------------------------
__global__ void conv_transpose_kernel(const float* __restrict__ src,
                                      unsigned short* __restrict__ dst,
                                      int K, int N, int Npad) {
    int idx = blockIdx.x * 256 + threadIdx.x;
    if (idx >= Npad * K) return;
    int n = idx / K, k = idx % K;
    float v = (n < N) ? src[k * N + n] : 0.0f;
    dst[idx] = f2bf(v);
}

__global__ void pad_bias_kernel(const float* __restrict__ src, float* __restrict__ dst,
                                int N, int Npad) {
    int i = blockIdx.x * 256 + threadIdx.x;
    if (i >= Npad) return;
    dst[i] = (i < N) ? src[i] : 0.0f;
}

// ---------------------------------------------------------------------------
// Fused per-row encoder: gather-embedder -> WMMA emb GEMM (LDS resident) ->
// 4x (attention softmax + LSTM cell) -> encoded[B,128] bf16.
// One workgroup (8 wave32) per batch row.
// ---------------------------------------------------------------------------
__global__ __launch_bounds__(256)
void encoder_kernel(const float* __restrict__ state,
                    const int*   __restrict__ acquired,
                    const float* __restrict__ We1, const float* __restrict__ be1,
                    const unsigned short* __restrict__ We2t, const float* __restrict__ be2,
                    const float* __restrict__ Wih, const float* __restrict__ Whh,
                    const float* __restrict__ bih, const float* __restrict__ bhh,
                    unsigned short* __restrict__ enc)
{
    __shared__ unsigned short s_hid[32 * EH];      // 8 KB  (32-token slab)
    __shared__ unsigned short s_emb[NFEAT * EDIM]; // 32 KB (whole-row embeddings)
    __shared__ int   s_valid[NFEAT];
    __shared__ float s_w[NFEAT];
    __shared__ float s_gates[4 * HDIM];
    __shared__ float s_red[256];
    __shared__ float s_att[EDIM];
    __shared__ float s_qt[HDIM];
    __shared__ float s_ct[HDIM];
    __shared__ int   s_len;

    const int b    = blockIdx.x;
    const int tid  = threadIdx.x;
    const int lane = tid & 31;
    const int wave = tid >> 5;

    const int av = (acquired[b * NFEAT + tid] != 0) ? 1 : 0;
    s_valid[tid] = av;
    if (tid < HDIM) { s_qt[tid] = 0.0f; s_ct[tid] = 0.0f; }
    float cnt = block_sum256(s_red, tid, (float)av);
    if (tid == 0) s_len = (int)(cnt + 0.5f);
    __syncthreads();

    // ---- token embedder: 8 slabs of 32 tokens -----------------------------
    for (int it = 0; it < 8; ++it) {
        const int tl = tid >> 3;            // local token 0..31
        const int j0 = (tid & 7) << 4;      // 16 hidden dims per thread
        const int t  = it * 32 + tl;
        const float val = state[b * NFEAT + t];
        const int vld = s_valid[t];
        const float* wf = We1 + (t + 1) * EH;
        #pragma unroll
        for (int jj = 0; jj < 16; ++jj) {
            int j = j0 + jj;
            float h = fmaf(val, We1[j], be1[j]);
            if (vld) h += wf[j];
            s_hid[tl * EH + j] = f2bf(fmaxf(h, 0.0f));
        }
        __syncthreads();

        // 8 waves: 2 M-tiles x 4 N-tiles; K = 128 in 4 WMMA steps
        const int mt = wave >> 2, nt = wave & 3;
        v8f acc;
        #pragma unroll
        for (int i = 0; i < 8; ++i) acc[i] = 0.0f;
        #pragma unroll
        for (int kb = 0; kb < EH; kb += 32) {
            v16bf a  = load_frag_a(s_hid, EH, mt * 16, kb, lane);
            v16bf bb = load_frag_b(We2t, EH, nt * 16, kb, lane);
            acc = __builtin_amdgcn_wmma_f32_16x16x32_bf16(false, a, false, bb,
                                                          (short)0, acc, false, false);
        }
        const int c     = nt * 16 + (lane & 15);
        const int rbase = it * 32 + mt * 16 + ((lane & 16) ? 8 : 0);
        const float bias = be2[c];
        #pragma unroll
        for (int j = 0; j < 8; ++j)
            s_emb[(rbase + j) * EDIM + c] = f2bf(acc[j] + bias);
        __syncthreads();
    }

    // ---- 4 rounds of masked attention + LSTM cell -------------------------
    for (int step = 0; step < 4; ++step) {
        float lg = -1e9f;
        if (s_valid[tid]) {
            float a = 0.0f;
            #pragma unroll 8
            for (int e = 0; e < EDIM; ++e)
                a += bf2f(s_emb[tid * EDIM + e]) * s_qt[e];
            lg = a;
        }
        float m  = block_max256(s_red, tid, lg);
        float ex = __expf(lg - m);
        s_w[tid] = ex;
        float sum = block_sum256(s_red, tid, ex);

        if (tid < EDIM) {
            float a = 0.0f;
            for (int t = 0; t < NFEAT; ++t)
                a += s_w[t] * bf2f(s_emb[t * EDIM + tid]);
            s_att[tid] = a / sum;
        }
        __syncthreads();

        // gates[256]: coalesced column reads of W_ih / W_hh
        {
            float g = bih[tid] + bhh[tid];
            #pragma unroll 8
            for (int k = 0; k < EDIM; ++k)
                g += s_att[k] * Wih[k * 256 + tid] + s_qt[k] * Whh[k * 256 + tid];
            s_gates[tid] = g;
        }
        __syncthreads();

        if (tid < HDIM) {
            float ig = sigmoidf(s_gates[tid]);
            float fg = sigmoidf(s_gates[HDIM + tid]);
            float gg = tanhf(s_gates[2 * HDIM + tid]);
            float og = sigmoidf(s_gates[3 * HDIM + tid]);
            float c  = fg * s_ct[tid] + ig * gg;
            s_ct[tid] = c;
            s_qt[tid] = og * tanhf(c);
        }
        __syncthreads();
    }

    // encoded = [attended || qt], zeroed when no acquired features
    if (tid < 2 * HDIM) {
        float v = (tid < EDIM) ? s_att[tid] : s_qt[tid - EDIM];
        if (s_len == 0) v = 0.0f;
        enc[b * 128 + tid] = f2bf(v);
    }
}

// ---------------------------------------------------------------------------
// Generic bf16 WMMA GEMM: C[M,N] = act(A[M,K] @ Bt^T + bias). 1 wave = 1 tile.
// ---------------------------------------------------------------------------
template<bool RELU, bool OUTF32>
__global__ __launch_bounds__(256)
void gemm_bf16_kernel(const unsigned short* __restrict__ A,
                      const unsigned short* __restrict__ Bt,
                      const float* __restrict__ bias,
                      void* __restrict__ Cout,
                      int N, int K, int tilesN, int numTiles)
{
    const int gt = (blockIdx.x * 256 + threadIdx.x) >> 5;
    if (gt >= numTiles) return;                       // wave-uniform exit
    const int lane = threadIdx.x & 31;
    const int tm = gt / tilesN, tn = gt % tilesN;

    v8f acc;
    #pragma unroll
    for (int i = 0; i < 8; ++i) acc[i] = 0.0f;
    for (int kb = 0; kb < K; kb += 32) {
        v16bf a  = load_frag_a(A,  K, tm * 16, kb, lane);
        v16bf bb = load_frag_b(Bt, K, tn * 16, kb, lane);
        acc = __builtin_amdgcn_wmma_f32_16x16x32_bf16(false, a, false, bb,
                                                      (short)0, acc, false, false);
    }
    const int c  = tn * 16 + (lane & 15);
    const int r0 = tm * 16 + ((lane & 16) ? 8 : 0);
    const float bv = bias[c];
    #pragma unroll
    for (int j = 0; j < 8; ++j) {
        float f = acc[j] + bv;
        if (RELU) f = fmaxf(f, 0.0f);
        if (OUTF32) ((float*)Cout)[(r0 + j) * N + c] = f;
        else        ((unsigned short*)Cout)[(r0 + j) * N + c] = f2bf(f);
    }
}

// ---------------------------------------------------------------------------
// Final head: v = V1 @ W_v2 + b_v2 ; out = v + adv - mean(adv[0:257]).
// ---------------------------------------------------------------------------
__global__ __launch_bounds__(256)
void head_final_kernel(const unsigned short* __restrict__ V1,
                       const float* __restrict__ Wv2, const float* __restrict__ bv2,
                       const float* __restrict__ ADV, float* __restrict__ out)
{
    __shared__ float red[256];
    const int b = blockIdx.x, tid = threadIdx.x;

    float pv = (tid < SDIM) ? bf2f(V1[b * SDIM + tid]) * Wv2[tid] : 0.0f;
    red[tid] = pv; __syncthreads();
    for (int s = 128; s >= 1; s >>= 1) { if (tid < s) red[tid] += red[tid + s]; __syncthreads(); }
    const float v = red[0] + bv2[0];
    __syncthreads();

    const float advt   = ADV[b * NAPAD + tid];
    const float adv256 = ADV[b * NAPAD + 256];
    red[tid] = advt + (tid == 0 ? adv256 : 0.0f); __syncthreads();
    for (int s = 128; s >= 1; s >>= 1) { if (tid < s) red[tid] += red[tid + s]; __syncthreads(); }
    const float mean = red[0] * (1.0f / 257.0f);

    out[b * NA + tid] = v + advt - mean;
    if (tid == 0) out[b * NA + 256] = v + adv256 - mean;
}

// ---------------------------------------------------------------------------
extern "C" void kernel_launch(void* const* d_in, const int* in_sizes, int n_in,
                              void* d_out, int out_size, void* d_ws, size_t ws_size,
                              hipStream_t stream)
{
    (void)in_sizes; (void)n_in; (void)out_size; (void)ws_size;
    const float* state = (const float*)d_in[0];
    const int*   acq   = (const int*)  d_in[1];
    const float* We1 = (const float*)d_in[2];  const float* be1 = (const float*)d_in[3];
    const float* We2 = (const float*)d_in[4];  const float* be2 = (const float*)d_in[5];
    const float* Wih = (const float*)d_in[6];  const float* Whh = (const float*)d_in[7];
    const float* bih = (const float*)d_in[8];  const float* bhh = (const float*)d_in[9];
    const float* Ws1 = (const float*)d_in[10]; const float* bs1 = (const float*)d_in[11];
    const float* Ws2 = (const float*)d_in[12]; const float* bs2 = (const float*)d_in[13];
    const float* Wp1 = (const float*)d_in[14]; const float* bp1 = (const float*)d_in[15];
    const float* Wp2 = (const float*)d_in[16]; const float* bp2 = (const float*)d_in[17];
    const float* Wv1 = (const float*)d_in[18]; const float* bv1 = (const float*)d_in[19];
    const float* Wv2 = (const float*)d_in[20]; const float* bv2 = (const float*)d_in[21];
    float* out = (float*)d_out;

    // workspace carve-up (~5.6 MB total)
    char* ws = (char*)d_ws; size_t off = 0;
    auto carve = [&](size_t bytes) -> char* {
        char* p = ws + off; off = (off + bytes + 255) & ~(size_t)255; return p;
    };
    unsigned short* We2t = (unsigned short*)carve(EDIM  * EH   * 2);
    unsigned short* Ws1t = (unsigned short*)carve(SHID  * 128  * 2);
    unsigned short* Ws2t = (unsigned short*)carve(SDIM  * SHID * 2);
    unsigned short* Wp1t = (unsigned short*)carve(SDIM  * SDIM * 2);
    unsigned short* Wp2t = (unsigned short*)carve(NAPAD * SDIM * 2);
    unsigned short* Wv1t = (unsigned short*)carve(SDIM  * SDIM * 2);
    float*          bp2p = (float*)         carve(NAPAD * 4);
    unsigned short* ENC  = (unsigned short*)carve((size_t)BATCH * 128  * 2);
    unsigned short* X1   = (unsigned short*)carve((size_t)BATCH * SHID * 2);
    unsigned short* SHB  = (unsigned short*)carve((size_t)BATCH * SDIM * 2);
    unsigned short* A1   = (unsigned short*)carve((size_t)BATCH * SDIM * 2);
    unsigned short* V1   = (unsigned short*)carve((size_t)BATCH * SDIM * 2);
    float*          ADV  = (float*)         carve((size_t)BATCH * NAPAD * 4);

    auto gridFor = [](int total) { return (total + 255) / 256; };

    // weight prep (bf16, transposed to [N,K]; W_p2 padded 257 -> 272 cols)
    conv_transpose_kernel<<<gridFor(EDIM  * EH  ), 256, 0, stream>>>(We2, We2t, EH,   EDIM, EDIM);
    conv_transpose_kernel<<<gridFor(SHID  * 128 ), 256, 0, stream>>>(Ws1, Ws1t, 128,  SHID, SHID);
    conv_transpose_kernel<<<gridFor(SDIM  * SHID), 256, 0, stream>>>(Ws2, Ws2t, SHID, SDIM, SDIM);
    conv_transpose_kernel<<<gridFor(SDIM  * SDIM), 256, 0, stream>>>(Wp1, Wp1t, SDIM, SDIM, SDIM);
    conv_transpose_kernel<<<gridFor(NAPAD * SDIM), 256, 0, stream>>>(Wp2, Wp2t, SDIM, NA,   NAPAD);
    conv_transpose_kernel<<<gridFor(SDIM  * SDIM), 256, 0, stream>>>(Wv1, Wv1t, SDIM, SDIM, SDIM);
    pad_bias_kernel<<<gridFor(NAPAD), 256, 0, stream>>>(bp2, bp2p, NA, NAPAD);

    // fused set-encoder (embedder GEMM + attention + LSTM), 1 WG per row
    encoder_kernel<<<BATCH, 256, 0, stream>>>(state, acq, We1, be1, We2t, be2,
                                              Wih, Whh, bih, bhh, ENC);

    // dueling head GEMM chain (wave-per-16x16-tile WMMA)
    auto launch_gemm = [&](auto kern, const unsigned short* A, const unsigned short* Bt,
                           const float* bias, void* C, int M, int N, int K) {
        int tilesN = N / 16, numTiles = (M / 16) * tilesN;
        kern<<<gridFor(numTiles * 32), 256, 0, stream>>>(A, Bt, bias, C, N, K, tilesN, numTiles);
    };
    launch_gemm(gemm_bf16_kernel<true,  false>, ENC, Ws1t, bs1,  X1,  BATCH, SHID,  128);
    launch_gemm(gemm_bf16_kernel<true,  false>, X1,  Ws2t, bs2,  SHB, BATCH, SDIM,  SHID);
    launch_gemm(gemm_bf16_kernel<true,  false>, SHB, Wp1t, bp1,  A1,  BATCH, SDIM,  SDIM);
    launch_gemm(gemm_bf16_kernel<false, true >, A1,  Wp2t, bp2p, ADV, BATCH, NAPAD, SDIM);
    launch_gemm(gemm_bf16_kernel<true,  false>, SHB, Wv1t, bv1,  V1,  BATCH, SDIM,  SDIM);

    head_final_kernel<<<BATCH, 256, 0, stream>>>(V1, Wv2, bv2, ADV, out);
}
